// MaskedPiecewiseRationalQuadraticAutoregressive_4870492913671
// MI455X (gfx1250) — matrix operand order repaired and checked by reading, not compiled, not run on hardware
//
#include <hip/hip_runtime.h>
#include <cstdint>
#include <cstddef>

// ---------------- problem constants ----------------
constexpr int Bq   = 16384;
constexpr int Dd   = 128;
constexpr int Hh   = 1024;
constexpr int Kb   = 8;          // spline bins
constexpr int OPD  = 3 * Kb + 1; // 25
constexpr int DOPD = Dd * OPD;   // 3200

typedef __attribute__((ext_vector_type(16))) __bf16 v16bf;
typedef __attribute__((ext_vector_type(8)))  __bf16 v8bf;
typedef __attribute__((ext_vector_type(8)))  float  v8f;

#if defined(__gfx1250__)
typedef int v4i __attribute__((vector_size(16)));
typedef __attribute__((address_space(1))) v4i GlobalV4i;  // global b128 payload
typedef __attribute__((address_space(3))) v4i LocalV4i;   // LDS b128 payload
#endif

// ---- 16-byte global -> LDS copy, async (CDNA5 GLOBAL_LOAD_ASYNC_TO_LDS_B128)
__device__ inline void cp_async16(void* lds_dst, const void* gsrc)
{
#if defined(__gfx1250__) && __has_builtin(__builtin_amdgcn_global_load_async_to_lds_b128)
    __builtin_amdgcn_global_load_async_to_lds_b128(
        (GlobalV4i*)(const_cast<void*>(gsrc)), (LocalV4i*)lds_dst, 0, 0);
#else
    *(v8bf*)lds_dst = *(const v8bf*)gsrc;  // fallback: vmem load + ds_store
#endif
}

template <int N>
__device__ inline void wait_asynccnt()
{
#if defined(__gfx1250__) && __has_builtin(__builtin_amdgcn_s_wait_asynccnt)
    __builtin_amdgcn_s_wait_asynccnt((unsigned short)N);
#elif defined(__gfx1250__)
    asm volatile("s_wait_asynccnt %0" :: "i"(N) : "memory");
#endif
}

// ---------------- pack: float (optionally masked) -> bf16 ----------------
__global__ void pack_bf16(const float* __restrict__ src,
                          const float* __restrict__ mask,
                          __bf16* __restrict__ dst, int n)
{
    int i = blockIdx.x * blockDim.x + threadIdx.x;
    if (i < n) {
        float v = src[i];
        if (mask) v *= mask[i];
        dst[i] = (__bf16)v;
    }
}

// ---------------- WMMA GEMM: C = op(A[M,K0] * W[N,K0]^T + bias) ----------------
// Block: 256 threads = 8 waves arranged 4(M) x 2(N).
// Wave tile: 32(M) x 64(N) = 2x4 v_wmma_f32_16x16x32_bf16 accumulators.
// Block tile: 128(M) x 128(N); K-step 32, double-buffered async LDS staging.
// Requires M%128==0, N%128==0, K0%32==0.
constexpr int TROW = 40;  // padded LDS row stride in bf16 (80B -> bank-conflict-free)

template <bool SILU, typename OutT>
__global__ __launch_bounds__(256)
void gemm_wmma(const __bf16* __restrict__ A, const __bf16* __restrict__ W,
               const float* __restrict__ bias, OutT* __restrict__ Cout,
               int M, int N, int K0)
{
    __shared__ __align__(16) __bf16 sA[2 * 128 * TROW];  // 20 KB
    __shared__ __align__(16) __bf16 sB[2 * 128 * TROW];  // 20 KB

    const int tid  = threadIdx.x;
    const int lane = tid & 31;
    const int wave = tid >> 5;
    const int wr   = wave & 3;   // wave row (M), 4 waves
    const int wc   = wave >> 2;  // wave col (N), 2 waves
    const int lm   = lane & 15;  // row within A-frag / col within B,C-frag
    const int lh   = lane >> 4;  // half selector (K grouping)

    const int blockM = blockIdx.y * 128;
    const int blockN = blockIdx.x * 128;
    const __bf16* Ablk = A + (size_t)blockM * K0;
    const __bf16* Wblk = W + (size_t)blockN * K0;

    // stage one 128x32 tile of A and of W into LDS buffer `buf`
    auto stage = [&](int buf, int kk) {
        __bf16* dA = sA + buf * 128 * TROW;
        __bf16* dB = sB + buf * 128 * TROW;
#pragma unroll
        for (int s = 0; s < 2; ++s) {
            const int c   = tid + s * 256;   // 512 16B-chunks per tile
            const int row = c >> 2;
            const int q   = c & 3;
            const __bf16* ga = Ablk + (size_t)row * K0 + kk + q * 8;
            const __bf16* gw = Wblk + (size_t)row * K0 + kk + q * 8;
            cp_async16(dA + row * TROW + q * 8, ga);
            cp_async16(dB + row * TROW + q * 8, gw);
            __builtin_prefetch((const void*)(ga + 64), 0, 0);
            __builtin_prefetch((const void*)(gw + 64), 0, 0);
        }
    };

    v8f acc[2][4] = {};
    const int nk = K0 / 32;

    stage(0, 0);
    for (int ki = 0; ki < nk; ++ki) {
        const int buf = ki & 1;
        if (ki + 1 < nk) {
            stage(buf ^ 1, (ki + 1) * 32);
            wait_asynccnt<4>();   // newest batch (4 ops) may remain in flight
        } else {
            wait_asynccnt<0>();
        }
        __syncthreads();

        const __bf16* pA = sA + buf * 128 * TROW;
        const __bf16* pB = sB + buf * 128 * TROW;

        // A-frag per CDNA5 16-bit A layout: lane holds row (lane&15);
        // elems 0..7 -> K = lh*8 + 0..7, elems 8..15 -> K = lh*8 + 16 + 0..7
        v16bf af[2];
#pragma unroll
        for (int i = 0; i < 2; ++i) {
            const __bf16* p = pA + (size_t)(wr * 32 + i * 16 + lm) * TROW + lh * 8;
            ((v8bf*)&af[i])[0] = *(const v8bf*)(p);
            ((v8bf*)&af[i])[1] = *(const v8bf*)(p + 16);
        }
        // B-frag per CDNA5 16-bit B layout: lane holds column (lane&15);
        // elems 0..7 -> K = lh*8 + 0..7, elems 8..15 -> K = lh*8 + 16 + 0..7
        v16bf bfr[4];
#pragma unroll
        for (int j = 0; j < 4; ++j) {
            const __bf16* p = pB + (size_t)(wc * 64 + j * 16 + lm) * TROW + lh * 8;
            ((v8bf*)&bfr[j])[0] = *(const v8bf*)(p);
            ((v8bf*)&bfr[j])[1] = *(const v8bf*)(p + 16);
        }

#pragma unroll
        for (int i = 0; i < 2; ++i)
#pragma unroll
            for (int j = 0; j < 4; ++j)
                acc[i][j] = __builtin_amdgcn_wmma_f32_16x16x32_bf16(
                    /*neg_a=*/false, af[i], /*neg_b=*/false, bfr[j],
                    /*c_mod=*/(short)0, acc[i][j],
                    /*reuse_a=*/false, /*reuse_b=*/false);

        __syncthreads();  // all waves done reading buf before it is re-staged
    }

    // Epilogue. C/D layout: lane holds column (lane&15); VGPR v -> row lh*8+v.
#pragma unroll
    for (int j = 0; j < 4; ++j) {
        const int gn = blockN + wc * 64 + j * 16 + lm;
        const float bv = bias[gn];
#pragma unroll
        for (int i = 0; i < 2; ++i) {
#pragma unroll
            for (int v = 0; v < 8; ++v) {
                const int gm = blockM + wr * 32 + i * 16 + lh * 8 + v;
                float val = acc[i][j][v] + bv;
                if (SILU) val = val / (1.0f + __expf(-val));
                Cout[(size_t)gm * N + gn] = (OutT)val;
            }
        }
    }
}

// ---------------- spline epilogue ----------------
__device__ inline void norm_cum9(const float* u, float mn, float* c /*[9]*/)
{
    float m = u[0];
#pragma unroll
    for (int k = 1; k < 8; ++k) m = fmaxf(m, u[k]);
    float e[8], s = 0.f;
#pragma unroll
    for (int k = 0; k < 8; ++k) { e[k] = __expf(u[k] - m); s += e[k]; }
    const float inv   = 1.0f / s;
    const float scale = 1.0f - mn * 8.0f;
    c[0] = 0.f;
    float a = 0.f;
#pragma unroll
    for (int k = 0; k < 8; ++k) { a += mn + scale * e[k] * inv; c[k + 1] = a; }
    const float dinv = 1.0f / fmaxf(c[8], 1e-12f);
#pragma unroll
    for (int k = 1; k <= 8; ++k) c[k] *= dinv;
}

__device__ inline float softplusf(float v)
{
    return (v > 20.f) ? v : log1pf(__expf(v));
}

__global__ __launch_bounds__(128)
void spline_kernel(const float* __restrict__ x, const float* __restrict__ raw,
                   float* __restrict__ out, float* __restrict__ lad)
{
    const int b = blockIdx.x;
    const int d = threadIdx.x; // 0..127
    const float xv = x[(size_t)b * Dd + d];
    const float* r = raw + (size_t)b * DOPD + d * OPD;

    float uw[8], uh[8], dd[9];
#pragma unroll
    for (int k = 0; k < 8; ++k) uw[k] = r[k];
#pragma unroll
    for (int k = 0; k < 8; ++k) uh[k] = r[8 + k];
#pragma unroll
    for (int k = 0; k < 9; ++k) dd[k] = 0.001f + softplusf(r[16 + k]);

    float cw[9], ch[9];
    norm_cum9(uw, 0.001f, cw);
    norm_cum9(uh, 0.001f, ch);

    const bool inside = (xv >= -3.f) && (xv <= 3.f);
    const float xs = fminf(fmaxf((xv + 3.f) / 6.f, 0.f), 1.f);

    int cnt = 0;
#pragma unroll
    for (int k = 0; k < 9; ++k) cnt += (xs >= cw[k]) ? 1 : 0;
    int bid = cnt - 1;
    bid = bid < 0 ? 0 : (bid > 7 ? 7 : bid);

    float xk = 0.f, wk = 0.f, yk = 0.f, hk = 0.f, dk = 0.f, dk1 = 0.f;
#pragma unroll
    for (int k = 0; k < 8; ++k) {
        if (k == bid) {
            xk = cw[k]; wk = cw[k + 1] - cw[k];
            yk = ch[k]; hk = ch[k + 1] - ch[k];
            dk = dd[k]; dk1 = dd[k + 1];
        }
    }

    const float EPSf = 1e-12f;
    float t   = fminf(fmaxf((xs - xk) / (wk + EPSf), 0.f), 1.f);
    float a   = (hk + EPSf) / (wk + EPSf);
    float omt = 1.f - t;
    float num = a * t * t + dk * t * omt;
    float den = a + (dk + dk1 - 2.f * a) * t * omt;
    float s   = num / (den + EPSf);
    float y   = (yk + hk * s) * 6.f - 3.f;
    float dnum = a * a * (dk1 * t * t + 2.f * a * t * omt + dk * omt * omt);
    float dydx = dnum / (den * den + EPSf);

    const float outv = inside ? y : xv;
    const float ladt = inside ? logf(fmaxf(dydx, 1e-12f)) : 0.f;

    out[(size_t)b * Dd + d] = outv;

    __shared__ float red[128];
    red[d] = ladt;
    __syncthreads();
#pragma unroll
    for (int sfac = 64; sfac > 0; sfac >>= 1) {
        if (d < sfac) red[d] += red[d + sfac];
        __syncthreads();
    }
    if (d == 0) lad[b] = red[0];
}

// ---------------- launcher ----------------
extern "C" void kernel_launch(void* const* d_in, const int* in_sizes, int n_in,
                              void* d_out, int out_size, void* d_ws, size_t ws_size,
                              hipStream_t stream)
{
    const float* x  = (const float*)d_in[0];
    const float* W1 = (const float*)d_in[1];
    const float* b1 = (const float*)d_in[2];
    const float* W2 = (const float*)d_in[3];
    const float* b2 = (const float*)d_in[4];
    const float* W3 = (const float*)d_in[5];
    const float* b3 = (const float*)d_in[6];
    const float* m1 = (const float*)d_in[7];
    const float* m2 = (const float*)d_in[8];
    const float* m3 = (const float*)d_in[9];

    char* ws = (char*)d_ws;
    size_t off = 0;
    auto carve = [&](size_t bytes) -> char* {
        char* p = ws + off;
        off = (off + bytes + 255) & ~(size_t)255;
        return p;
    };
    __bf16* xb  = (__bf16*)carve((size_t)Bq * Dd * 2);
    __bf16* w1b = (__bf16*)carve((size_t)Hh * Dd * 2);
    __bf16* w2b = (__bf16*)carve((size_t)Hh * Hh * 2);
    __bf16* w3b = (__bf16*)carve((size_t)DOPD * Hh * 2);
    __bf16* h1  = (__bf16*)carve((size_t)Bq * Hh * 2);
    __bf16* h2  = (__bf16*)carve((size_t)Bq * Hh * 2);
    float*  raw = (float*)carve((size_t)Bq * DOPD * 4);
    (void)ws_size;

    // ---- pack inputs / masked weights to bf16 ----
    {
        int n = Bq * Dd;
        pack_bf16<<<(n + 255) / 256, 256, 0, stream>>>(x, nullptr, xb, n);
        n = Hh * Dd;
        pack_bf16<<<(n + 255) / 256, 256, 0, stream>>>(W1, m1, w1b, n);
        n = Hh * Hh;
        pack_bf16<<<(n + 255) / 256, 256, 0, stream>>>(W2, m2, w2b, n);
        n = DOPD * Hh;
        pack_bf16<<<(n + 255) / 256, 256, 0, stream>>>(W3, m3, w3b, n);
    }

    // ---- GEMM 1: h1 = silu(x @ W1m^T + b1)  [16384 x 128 x 1024] ----
    {
        dim3 grid(Hh / 128, Bq / 128);
        gemm_wmma<true, __bf16><<<grid, 256, 0, stream>>>(xb, w1b, b1, h1, Bq, Hh, Dd);
    }
    // ---- GEMM 2: h2 = silu(h1 @ W2m^T + b2) [16384 x 1024 x 1024] ----
    {
        dim3 grid(Hh / 128, Bq / 128);
        gemm_wmma<true, __bf16><<<grid, 256, 0, stream>>>(h1, w2b, b2, h2, Bq, Hh, Hh);
    }
    // ---- GEMM 3: raw = h2 @ W3m^T + b3      [16384 x 1024 x 3200] ----
    {
        dim3 grid(DOPD / 128, Bq / 128);
        gemm_wmma<false, float><<<grid, 256, 0, stream>>>(h2, w3b, b3, raw, Bq, DOPD, Hh);
    }

    // ---- spline epilogue: out (B,D) fp32 + lad (B,) fp32, concatenated ----
    {
        float* outp = (float*)d_out;
        float* ladp = outp + (size_t)Bq * Dd;
        spline_kernel<<<Bq, 128, 0, stream>>>(x, raw, outp, ladp);
    }
}
